// MultiHeadAttention_87875030876842
// MI455X (gfx1250) — compile-verified, matrix operand order checked by reading, not run on hardware
//
#include <hip/hip_runtime.h>
#include <hip/hip_bf16.h>

typedef __attribute__((ext_vector_type(16))) _Float16 v16h;
typedef __attribute__((ext_vector_type(8)))  _Float16 v8h;
typedef __attribute__((ext_vector_type(8)))  float    v8f;

#define WMMA_F16(a, b, c) \
  __builtin_amdgcn_wmma_f32_16x16x32_f16(false, (a), false, (b), (short)0, (c), false, false)

static constexpr int Bsz = 2;
static constexpr int S   = 2048;
static constexpr int D   = 768;
static constexpr int H   = 12;
static constexpr int DK  = 64;
static constexpr int BS  = Bsz * S;   // 4096

// Generic 16-bit WMMA fragment loader (A- or B-fragment; layouts mirror).
// Lane l (row/col index m = l&15) holds K = {ko..ko+7, ko+16..ko+23}, ko = (l<16)?0:8.
// Two 16-byte loads per lane.
__device__ __forceinline__ v16h load_frag_f16(const _Float16* base, int stride) {
  const int lane = threadIdx.x & 31;
  const int m    = lane & 15;
  const int ko   = (lane < 16) ? 0 : 8;
  const _Float16* p = base + (size_t)m * stride + ko;
  v8h lo = *(const v8h*)(p);
  v8h hi = *(const v8h*)(p + 16);
  v16h r;
#pragma unroll
  for (int i = 0; i < 8; ++i) { r[i] = lo[i]; r[i + 8] = hi[i]; }
  return r;
}

// fp32 -> f16 convert
__global__ void k_cvt(const float* __restrict__ src, _Float16* __restrict__ dst, int n) {
  int i = blockIdx.x * 256 + threadIdx.x;
  if (i < n) dst[i] = (_Float16)src[i];
}

// Y = X @ W^T + bias, written f16 into head-split layouts.
// mode 0: Q -> [b,h,s,dk], scaled by 1/sqrt(dk)
// mode 1: K -> [b,h,s,dk]
// mode 2: V -> [b,h,dk,s]  (transposed so P@V B-fragments load contiguously)
__global__ void k_proj_qkv(const _Float16* __restrict__ X, const _Float16* __restrict__ W,
                           const float* __restrict__ bias, _Float16* __restrict__ dst, int mode) {
  const int tm = blockIdx.x;          // row tile over BS/16
  const int tn = blockIdx.y;          // col tile over D/16
  const int lane = threadIdx.x & 31;
  v8f c = {};
#pragma unroll 4
  for (int kc = 0; kc < D; kc += 32) {
    v16h a = load_frag_f16(X + (size_t)tm * 16 * D + kc, D);
    v16h b = load_frag_f16(W + (size_t)tn * 16 * D + kc, D);
    c = WMMA_F16(a, b, c);
  }
  const int n   = lane & 15;
  const int col = tn * 16 + n;
  const float bval = bias[col];
  const int hh = col >> 6;
  const int dd = col & 63;
  const float sc = (mode == 0) ? 0.125f : 1.0f;   // fold 1/sqrt(64) into Q
#pragma unroll
  for (int r = 0; r < 8; ++r) {
    int row = tm * 16 + r + ((lane < 16) ? 0 : 8);
    int bb  = row >> 11;              // row / S
    int ss  = row & (S - 1);
    float vv = (c[r] + bval) * sc;
    size_t off = (mode == 2)
        ? ((((size_t)(bb * H + hh)) * DK + dd) * S + ss)
        : ((((size_t)(bb * H + hh)) * S + ss) * DK + dd);
    dst[off] = (_Float16)vv;
  }
}

// Fused scores -> mask -> softmax -> attn write -> ctx = P@V, per 16-row query tile.
// One wave per block; 64 KB dynamic LDS holds the 16 x 2048 probability row (f16).
__global__ void k_attn(const _Float16* __restrict__ Qh, const _Float16* __restrict__ Kh,
                       const _Float16* __restrict__ Vt, const int* __restrict__ mask,
                       float* __restrict__ attn, _Float16* __restrict__ ctx) {
  extern __shared__ _Float16 P[];     // [16][S]
  const int lane = threadIdx.x & 31;
  const int qt = blockIdx.x;          // 0..S/16-1
  const int bh = blockIdx.y;          // 0..B*H-1
  const int bb = bh / H;
  const int hh = bh % H;
  const int q0 = qt * 16;
  const _Float16* Qb = Qh + (size_t)bh * S * DK + (size_t)q0 * DK;
  const _Float16* Kb = Kh + (size_t)bh * S * DK;
  const _Float16* Vb = Vt + (size_t)bh * DK * S;

  // persistent Q A-fragments (dk=64 -> two 16x32 chunks)
  const v16h aq0 = load_frag_f16(Qb + 0,  DK);
  const v16h aq1 = load_frag_f16(Qb + 32, DK);

  const int n    = lane & 15;
  const int rofs = (lane < 16) ? 0 : 8;

  // ---- scores = Q @ K^T (pre-scaled), masked, -> LDS f16 ----
  for (int kt = 0; kt < S / 16; ++kt) {
    v8f c = {};
    v16h b0 = load_frag_f16(Kb + (size_t)kt * 16 * DK + 0,  DK);
    v16h b1 = load_frag_f16(Kb + (size_t)kt * 16 * DK + 32, DK);
    c = WMMA_F16(aq0, b0, c);
    c = WMMA_F16(aq1, b1, c);
    const int col = kt * 16 + n;
#pragma unroll
    for (int r = 0; r < 8; ++r) {
      int row = r + rofs;
      int mv  = mask[(size_t)(q0 + row) * S + col];
      float s = (mv == 0) ? -30000.0f : c[r];   // exp(-30000-mx) == 0 in f32, matches -1e9 path
      P[row * S + col] = (_Float16)s;
    }
  }
  __syncthreads();

  // ---- softmax: 2 lanes per row, f32 statistics over f16 scores ----
  {
    const int row = lane >> 1;
    const int c0  = (lane & 1) * (S / 2);
    _Float16* pr = P + row * S + c0;
    float mx = -3.0e38f;
    for (int j = 0; j < S / 2; j += 8) {
      v8h v = *(const v8h*)(pr + j);
#pragma unroll
      for (int i = 0; i < 8; ++i) mx = fmaxf(mx, (float)v[i]);
    }
    mx = fmaxf(mx, __shfl_xor(mx, 1, 32));
    float sum = 0.0f;
    for (int j = 0; j < S / 2; j += 8) {
      v8h v = *(const v8h*)(pr + j);
      v8h e;
#pragma unroll
      for (int i = 0; i < 8; ++i) {
        float t = __expf((float)v[i] - mx);
        sum += t;
        e[i] = (_Float16)t;
      }
      *(v8h*)(pr + j) = e;
    }
    sum += __shfl_xor(sum, 1, 32);
    float inv = 1.0f / sum;
    for (int j = 0; j < S / 2; j += 8) {
      v8h v = *(const v8h*)(pr + j);
#pragma unroll
      for (int i = 0; i < 8; ++i) v[i] = (_Float16)((float)v[i] * inv);
      *(v8h*)(pr + j) = v;
    }
  }
  __syncthreads();

  // ---- write attn (fp32, coalesced float4; rows q0..q0+15 are contiguous) ----
  {
    float* arow = attn + ((size_t)bh * S + q0) * S;
    for (int i = lane; i < 16 * S / 4; i += 32) {
      int idx = i * 4;
      float4 v = { (float)P[idx], (float)P[idx + 1], (float)P[idx + 2], (float)P[idx + 3] };
      *(float4*)(arow + idx) = v;
    }
  }

  // ---- ctx = P @ V  (A from LDS probs, B from transposed V) ----
  v8f cc0 = {}, cc1 = {}, cc2 = {}, cc3 = {};
  for (int st = 0; st < S; st += 32) {
    v16h a  = load_frag_f16(P + st, S);
    v16h b0 = load_frag_f16(Vb + (size_t)0 * 16 * S + st, S);
    v16h b1 = load_frag_f16(Vb + (size_t)1 * 16 * S + st, S);
    v16h b2 = load_frag_f16(Vb + (size_t)2 * 16 * S + st, S);
    v16h b3 = load_frag_f16(Vb + (size_t)3 * 16 * S + st, S);
    cc0 = WMMA_F16(a, b0, cc0);
    cc1 = WMMA_F16(a, b1, cc1);
    cc2 = WMMA_F16(a, b2, cc2);
    cc3 = WMMA_F16(a, b3, cc3);
  }
  {
    v8f ccs[4] = {cc0, cc1, cc2, cc3};
#pragma unroll
    for (int nt = 0; nt < 4; ++nt) {
#pragma unroll
      for (int r = 0; r < 8; ++r) {
        int rw  = q0 + r + rofs;
        int col = hh * DK + nt * 16 + n;
        ctx[((size_t)bb * S + rw) * D + col] = (_Float16)ccs[nt][r];
      }
    }
  }
}

// out = ctx @ Wo^T + bo (fp32 result)
__global__ void k_proj_out(const _Float16* __restrict__ X, const _Float16* __restrict__ W,
                           const float* __restrict__ bias, float* __restrict__ out) {
  const int tm = blockIdx.x;
  const int tn = blockIdx.y;
  const int lane = threadIdx.x & 31;
  v8f c = {};
#pragma unroll 4
  for (int kc = 0; kc < D; kc += 32) {
    v16h a = load_frag_f16(X + (size_t)tm * 16 * D + kc, D);
    v16h b = load_frag_f16(W + (size_t)tn * 16 * D + kc, D);
    c = WMMA_F16(a, b, c);
  }
  const int col = tn * 16 + (lane & 15);
  const float bval = bias[col];
#pragma unroll
  for (int r = 0; r < 8; ++r) {
    int row = tm * 16 + r + ((lane < 16) ? 0 : 8);
    out[(size_t)row * D + col] = c[r] + bval;
  }
}

extern "C" void kernel_launch(void* const* d_in, const int* in_sizes, int n_in,
                              void* d_out, int out_size, void* d_ws, size_t ws_size,
                              hipStream_t stream) {
  (void)in_sizes; (void)n_in; (void)out_size; (void)ws_size;
  const float* q    = (const float*)d_in[0];
  const float* k    = (const float*)d_in[1];
  const float* v    = (const float*)d_in[2];
  const int*   mask = (const int*)d_in[3];
  const float* wq   = (const float*)d_in[4];
  const float* bq   = (const float*)d_in[5];
  const float* wk   = (const float*)d_in[6];
  const float* bk   = (const float*)d_in[7];
  const float* wv   = (const float*)d_in[8];
  const float* bvp  = (const float*)d_in[9];
  const float* wo   = (const float*)d_in[10];
  const float* bo   = (const float*)d_in[11];

  float* out  = (float*)d_out;                 // [B,S,D]
  float* attn = out + (size_t)BS * D;          // [B,H,S,S]

  _Float16* w = (_Float16*)d_ws;
  _Float16* Xq  = w; w += (size_t)BS * D;
  _Float16* Xk  = w; w += (size_t)BS * D;
  _Float16* Xv  = w; w += (size_t)BS * D;
  _Float16* Wq  = w; w += (size_t)D * D;
  _Float16* Wk  = w; w += (size_t)D * D;
  _Float16* Wv  = w; w += (size_t)D * D;
  _Float16* Wo  = w; w += (size_t)D * D;
  _Float16* Qh  = w; w += (size_t)BS * D;      // [b,h,s,dk]
  _Float16* Kh  = w; w += (size_t)BS * D;      // [b,h,s,dk]
  _Float16* Vt  = w; w += (size_t)BS * D;      // [b,h,dk,s]
  _Float16* Ctx = w; w += (size_t)BS * D;      // [b,s,d]

  const int nX = BS * D;
  const int nW = D * D;
  k_cvt<<<(nX + 255) / 256, 256, 0, stream>>>(q,  Xq, nX);
  k_cvt<<<(nX + 255) / 256, 256, 0, stream>>>(k,  Xk, nX);
  k_cvt<<<(nX + 255) / 256, 256, 0, stream>>>(v,  Xv, nX);
  k_cvt<<<(nW + 255) / 256, 256, 0, stream>>>(wq, Wq, nW);
  k_cvt<<<(nW + 255) / 256, 256, 0, stream>>>(wk, Wk, nW);
  k_cvt<<<(nW + 255) / 256, 256, 0, stream>>>(wv, Wv, nW);
  k_cvt<<<(nW + 255) / 256, 256, 0, stream>>>(wo, Wo, nW);

  dim3 gp(BS / 16, D / 16);
  k_proj_qkv<<<gp, 32, 0, stream>>>(Xq, Wq, bq,  Qh, 0);
  k_proj_qkv<<<gp, 32, 0, stream>>>(Xk, Wk, bk,  Kh, 1);
  k_proj_qkv<<<gp, 32, 0, stream>>>(Xv, Wv, bvp, Vt, 2);

  dim3 ga(S / 16, Bsz * H);
  k_attn<<<ga, 32, (size_t)16 * S * sizeof(_Float16), stream>>>(Qh, Kh, Vt, mask, attn, Ctx);

  k_proj_out<<<gp, 32, 0, stream>>>(Ctx, Wo, bo, out);
}
